// Enformer_25555055411595
// MI455X (gfx1250) — compile-verified, matrix operand order checked by reading, not run on hardware
//
#include <hip/hip_runtime.h>
#include <cstdint>
#include <cstddef>

typedef _Float16 f16;
typedef long long i64;
typedef __attribute__((ext_vector_type(16))) _Float16 v16h;
typedef __attribute__((ext_vector_type(8)))  _Float16 v8h;
typedef __attribute__((ext_vector_type(8)))  float    v8f;

#define DEV __device__ __forceinline__

DEV float gelu_f(float x){ float x3=x*x*x; return 0.5f*x*(1.f+tanhf(0.7978845608f*(x+0.044715f*x3))); }
DEV float bn_f(float x,float g,float b,float m,float v){ return (x-m)*rsqrtf(v+1e-3f)*g+b; }

// ---- CDNA5 async global->LDS staging (32 bytes: two b128 ops), ASYNCcnt tracked ----
DEV void async_copy32(const f16* gsrc, f16* lds_dst){
  unsigned l0 = (unsigned)(uintptr_t)lds_dst;          // generic LDS addr[31:0] == LDS byte offset
  unsigned long long ga = (unsigned long long)(uintptr_t)gsrc;
  asm volatile("global_load_async_to_lds_b128 %0, %1, off\n\t"
               "global_load_async_to_lds_b128 %2, %1, off offset:16"
               :: "v"(l0), "v"(ga), "v"(l0+16u) : "memory");
}
DEV void async_wait0(){
  asm volatile("s_wait_asynccnt 0" ::: "memory");
}

// =============== WMMA GEMM: C = act(alpha*A@B + bias + residual) ===============
// A: f16 [M,K] row-major (lda, 16B-aligned rows, K%32==0)
// B: f16 [K,N] (ldb) or, if transB, [N,K] (ldb)
// Outputs optionally f32 (Cf) and/or f16 (Ch), both ldc. grid.z = batch.
__global__ __launch_bounds__(256) void k_gemm(
    const f16* __restrict__ A, int lda, i64 sA,
    const f16* __restrict__ B, int ldb, i64 sB, int transB,
    const float* __restrict__ bias,
    const float* __restrict__ resid, i64 sR,
    float* __restrict__ Cf, f16* __restrict__ Ch, int ldc, i64 sC,
    int M, int N, int K, float alpha, int act)
{
  __shared__ f16 As[128][32];
  __shared__ f16 Bs[128][32];   // [n][k]
  const int bz = blockIdx.z;
  A += (i64)bz*sA; B += (i64)bz*sB;
  const int m0 = blockIdx.y*128, n0 = blockIdx.x*128;
  const int tid = threadIdx.x, lane = tid&31, wid = tid>>5;
  const int wm = (wid&3)*32, wn = (wid>>2)*64;

  v8f zf = {0.f,0.f,0.f,0.f,0.f,0.f,0.f,0.f};
  v8f acc[2][4];
  #pragma unroll
  for(int i=0;i<2;++i)
    #pragma unroll
    for(int j=0;j<4;++j) acc[i][j]=zf;

  const int arow = tid>>1, acol=(tid&1)*16;
  // clamped row indices: out-of-range rows/cols land only in epilogue-masked tile
  int gm = m0+arow; if (gm >= M) gm = M-1;
  int gncl = n0+arow; if (gncl >= N) gncl = N-1;

  const int kTiles = K>>5;
  for(int kt=0;kt<kTiles;++kt){
    const int k0 = kt<<5;
    __syncthreads();   // all waves done reading previous tile
    // ---- stage A tile [128][32] via async HBM->LDS ----
    async_copy32(A + (i64)gm*lda + k0 + acol, &As[arow][acol]);
    if (transB){
      // ---- stage B tile from [N,K] via async ----
      async_copy32(B + (i64)gncl*ldb + k0 + acol, &Bs[arow][acol]);
    } else {
      // ---- stage B tile from [K,N], transpose into Bs[n][k] (sync path) ----
      int kk2 = tid & 31;
      int nb  = (tid >> 5) * 16;
      const f16* bp = B + (i64)(k0+kk2)*ldb + n0;
      if (((ldb & 7)==0) && (n0 + nb + 15) < N){
        v8h b0 = *(const v8h*)(bp + nb);
        v8h b1 = *(const v8h*)(bp + nb + 8);
        #pragma unroll
        for(int i=0;i<8;++i){ Bs[nb+i][kk2]=b0[i]; Bs[nb+8+i][kk2]=b1[i]; }
      } else {
        #pragma unroll
        for(int i=0;i<16;++i){
          int n = n0+nb+i; if (n >= N) n = N-1;
          Bs[nb+i][kk2] = bp[n-n0];
        }
      }
    }
    async_wait0();
    __syncthreads();
    // fragment loads per ISA 16-bit A layout: lane<16 -> K {0..7,16..23}; lane>=16 -> {8..15,24..31}
    const int kb = (lane<16)?0:8;
    const int lr = lane&15;
    v16h af[2], bf[4];
    #pragma unroll
    for(int mi=0;mi<2;++mi){
      union{v16h v; v8h h[2];}u;
      u.h[0]=*(const v8h*)&As[wm+mi*16+lr][kb];
      u.h[1]=*(const v8h*)&As[wm+mi*16+lr][kb+16];
      af[mi]=u.v;
    }
    #pragma unroll
    for(int ni=0;ni<4;++ni){
      union{v16h v; v8h h[2];}u;
      u.h[0]=*(const v8h*)&Bs[wn+ni*16+lr][kb];
      u.h[1]=*(const v8h*)&Bs[wn+ni*16+lr][kb+16];
      bf[ni]=u.v;
    }
    #pragma unroll
    for(int mi=0;mi<2;++mi)
      #pragma unroll
      for(int ni=0;ni<4;++ni)
        acc[mi][ni]=__builtin_amdgcn_wmma_f32_16x16x32_f16(
            false, af[mi], false, bf[ni], (short)0, acc[mi][ni], false, false);
  }
  // epilogue: lane<16 -> rows 0..7 (elem i = M=i), lane>=16 -> rows 8..15; col = lane&15
  const int cl = lane&15;
  const int ro = (lane&16)?8:0;
  float* CfB = Cf ? Cf + (i64)bz*sC : nullptr;
  f16*   ChB = Ch ? Ch + (i64)bz*sC : nullptr;
  const float* RsB = resid ? resid + (i64)bz*sR : nullptr;
  #pragma unroll
  for(int mi=0;mi<2;++mi)
    #pragma unroll
    for(int ni=0;ni<4;++ni){
      int col = n0+wn+ni*16+cl;
      if (col>=N) continue;
      float bv = bias ? bias[col] : 0.f;
      #pragma unroll
      for(int i=0;i<8;++i){
        int row = m0+wm+mi*16+ro+i;
        if (row>=M) continue;
        float v = acc[mi][ni][i]*alpha + bv;
        i64 off = (i64)row*ldc + col;
        if (RsB) v += RsB[off];
        if (act==1)      v = fmaxf(v,0.f);
        else if (act==2) v = gelu_f(v);
        else if (act==3) v = (v>20.f)? v : log1pf(__expf(v));
        if (CfB) CfB[off]=v;
        if (ChB) ChB[off]=(f16)v;
      }
    }
}

// =============== elementwise / reduction kernels ===============
__global__ __launch_bounds__(256) void k_f32_to_f16(const float* __restrict__ x, f16* __restrict__ y, i64 n){
  i64 i = (i64)blockIdx.x*256 + threadIdx.x;
  if (i<n) y[i]=(f16)x[i];
}
__global__ __launch_bounds__(256) void k_zero1(float* p){ if (threadIdx.x==0) p[0]=0.f; }

__global__ __launch_bounds__(256) void k_stem_conv(const float* __restrict__ x, const float* __restrict__ w,
                                                   const float* __restrict__ b, float* __restrict__ y,
                                                   int T, int Cout){
  i64 idx = (i64)blockIdx.x*256 + threadIdx.x;
  if (idx >= (i64)T*Cout) return;
  int c = (int)(idx % Cout); i64 t = idx / Cout;
  float acc = b[c];
  #pragma unroll
  for(int dw=0; dw<15; ++dw){
    i64 ti = t + dw - 7;
    if (ti < 0 || ti >= T) continue;
    const float* xp = x + ti*4;
    const float* wp = w + (i64)(dw*4)*Cout + c;
    acc += xp[0]*wp[0] + xp[1]*wp[Cout] + xp[2]*wp[2*(i64)Cout] + xp[3]*wp[3*(i64)Cout];
  }
  y[idx]=acc;
}

__global__ __launch_bounds__(256) void k_bn_gelu_f16(const float* __restrict__ x,
    const float* __restrict__ g, const float* __restrict__ b,
    const float* __restrict__ m, const float* __restrict__ v,
    f16* __restrict__ out, i64 n, int Cc){
  i64 i = (i64)blockIdx.x*256 + threadIdx.x;
  if (i>=n) return;
  int c = (int)(i % Cc);
  out[i] = (f16)gelu_f(bn_f(x[i], g[c], b[c], m[c], v[c]));
}

__global__ __launch_bounds__(256) void k_im2col_bn_gelu(const float* __restrict__ x,
    const float* __restrict__ g, const float* __restrict__ b,
    const float* __restrict__ m, const float* __restrict__ v,
    f16* __restrict__ out, int T, int cin){
  i64 n = (i64)T*5*cin;
  i64 i = (i64)blockIdx.x*256 + threadIdx.x;
  if (i>=n) return;
  int kw = 5*cin;
  int j = (int)(i % kw); i64 t = i / kw;
  int dw = j / cin, ci = j - dw*cin;
  i64 ti = t + dw - 2;
  float val = 0.f;
  if (ti>=0 && ti<T) val = gelu_f(bn_f(x[ti*(i64)cin+ci], g[ci], b[ci], m[ci], v[ci]));
  out[i] = (f16)val;
}

__global__ __launch_bounds__(256) void k_pool2(const float* __restrict__ x, float* __restrict__ out,
                                               int T2, int Cc){
  i64 i = (i64)blockIdx.x*256 + threadIdx.x;
  if (i >= (i64)T2*Cc) return;
  int c = (int)(i % Cc); i64 t2 = i / Cc;
  float a = x[(2*t2)*(i64)Cc + c], d = x[(2*t2+1)*(i64)Cc + c];
  float mx = fmaxf(a,d);
  float ea = __expf(a-mx), ed = __expf(d-mx);
  out[i] = (a*ea + d*ed) / (ea + ed);
}

__global__ __launch_bounds__(256) void k_layernorm_f16(const float* __restrict__ x,
    const float* __restrict__ g, const float* __restrict__ b,
    f16* __restrict__ out, int Cc){
  int row = blockIdx.x;
  const float* xp = x + (i64)row*Cc;
  __shared__ float red[256];
  float s=0.f;
  for(int c=threadIdx.x;c<Cc;c+=256) s+=xp[c];
  red[threadIdx.x]=s; __syncthreads();
  for(int o=128;o>0;o>>=1){ if(threadIdx.x<o) red[threadIdx.x]+=red[threadIdx.x+o]; __syncthreads(); }
  float mean = red[0]/Cc; __syncthreads();
  float vv=0.f;
  for(int c=threadIdx.x;c<Cc;c+=256){ float d=xp[c]-mean; vv+=d*d; }
  red[threadIdx.x]=vv; __syncthreads();
  for(int o=128;o>0;o>>=1){ if(threadIdx.x<o) red[threadIdx.x]+=red[threadIdx.x+o]; __syncthreads(); }
  float rstd = rsqrtf(red[0]/Cc + 1e-3f);
  for(int c=threadIdx.x;c<Cc;c+=256)
    out[(i64)row*Cc+c] = (f16)((xp[c]-mean)*rstd*g[c]+b[c]);
}

// out[t, h*64+k] = qs[t, h*64+k] + r[h,k]
__global__ __launch_bounds__(256) void k_add_head_bias(const float* __restrict__ qs,
    const float* __restrict__ r, f16* __restrict__ out, i64 n){
  i64 i = (i64)blockIdx.x*256 + threadIdx.x;
  if (i>=n) return;
  int j = (int)(i & 511);
  out[i] = (f16)(qs[i] + r[j]);   // r flat [8*64] matches j in [0,512)
}

// fused relative_shift gather + softmax; T must be 1024 here
__global__ __launch_bounds__(256) void k_attn_softmax(const float* __restrict__ content,
    const float* __restrict__ rel, f16* __restrict__ probs, int T){
  int q = blockIdx.x, h = blockIdx.y;
  const float* cp = content + ((i64)h*T + q)*(i64)T;
  const float* rp = rel + ((i64)h*T + q)*(i64)(2*T-1);
  f16* pp = probs + ((i64)h*T + q)*(i64)T;
  __shared__ float red[256];
  float sv[4];
  float mx = -1e30f;
  #pragma unroll
  for(int i=0;i<4;++i){
    int m = threadIdx.x + i*256;
    float s = cp[m] + rp[T-1+m-q];
    sv[i]=s; mx=fmaxf(mx,s);
  }
  red[threadIdx.x]=mx; __syncthreads();
  for(int o=128;o>0;o>>=1){ if(threadIdx.x<o) red[threadIdx.x]=fmaxf(red[threadIdx.x],red[threadIdx.x+o]); __syncthreads(); }
  mx = red[0]; __syncthreads();
  float se=0.f;
  #pragma unroll
  for(int i=0;i<4;++i){ float e=__expf(sv[i]-mx); sv[i]=e; se+=e; }
  red[threadIdx.x]=se; __syncthreads();
  for(int o=128;o>0;o>>=1){ if(threadIdx.x<o) red[threadIdx.x]+=red[threadIdx.x+o]; __syncthreads(); }
  float inv = 1.f/red[0];
  #pragma unroll
  for(int i=0;i<4;++i) pp[threadIdx.x + i*256] = (f16)(sv[i]*inv);
}

// =============== relative position features (T=1024, NB=32) ===============
DEV float lgamma_s(float z){
  float invz = 1.f/z;
  return (z-0.5f)*logf(z) - z + 0.9189385332f + invz*(0.08333333333f - 0.002777777778f*invz*invz);
}
__global__ __launch_bounds__(256) void k_pe_prob(float* __restrict__ prob, float* __restrict__ pmax, int T){
  int L = 2*T-1;
  int idx = blockIdx.x*256 + threadIdx.x;
  if (idx >= L*32) return;
  int l = idx >> 5, i = idx & 31;
  float pos = (float)(l - (T-1));
  float ap = fabsf(pos);
  float mean = 32.f*(i+1);           // linspace(32,1024,32)
  float conc = (mean/16.f)*(mean/16.f);
  float rate = mean/256.f;
  float lp = (conc-1.f)*logf(fmaxf(ap,1e-20f)) - rate*ap - lgamma_s(conc) + conc*logf(rate);
  float p = (ap==0.f ? 0.f : __expf(lp)) + 1e-8f;
  prob[idx]=p;
  atomicMax((unsigned int*)pmax, __float_as_uint(p));
}
__global__ __launch_bounds__(256) void k_pe_emb(const float* __restrict__ prob, const float* __restrict__ pmax,
                                                f16* __restrict__ pe, int T){
  int L = 2*T-1;
  i64 idx = (i64)blockIdx.x*256 + threadIdx.x;
  if (idx >= (i64)L*192) return;
  int l = (int)(idx / 192), e = (int)(idx % 192);
  float pos = (float)(l - (T-1));
  float ap = fabsf(pos);
  float sgn = (e < 96) ? 1.f : ((pos>0.f)?1.f:((pos<0.f)?-1.f:0.f));
  int ee = e % 96;
  float val;
  if (ee < 32){
    float hl = exp2f(3.f + 7.f*ee/31.f);
    val = __expf(-0.6931471806f/hl * ap);
  } else if (ee < 64){
    int i = ee-32;
    val = ((exp2f((float)(i+1)) - 1.f) > ap) ? 1.f : 0.f;
  } else {
    val = prob[l*32 + (ee-64)] / pmax[0];
  }
  pe[idx] = (f16)(sgn*val);
}

// =============== host orchestration ===============
struct BNp { const float *g,*b,*m,*v; };
struct CBp { BNp bn; const float *w, *bias; };

static inline dim3 g1(i64 n){ return dim3((unsigned)((n+255)/256)); }

static void gemm(hipStream_t s,
                 const f16* A,int lda,i64 sA,
                 const f16* B,int ldb,i64 sB,int transB,
                 const float* bias, const float* resid, i64 sR,
                 float* Cf, f16* Ch, int ldc, i64 sC,
                 int M,int N,int K,float alpha,int act,int batch){
  dim3 grid((N+127)/128, (M+127)/128, batch);
  k_gemm<<<grid, 256, 0, s>>>(A,lda,sA,B,ldb,sB,transB,bias,resid,sR,Cf,Ch,ldc,sC,M,N,K,alpha,act);
}

extern "C" void kernel_launch(void* const* d_in, const int* in_sizes, int n_in,
                              void* d_out, int out_size, void* d_ws, size_t ws_size,
                              hipStream_t stream) {
  (void)in_sizes; (void)n_in; (void)out_size; (void)ws_size;
  const int SEQ=65536, C0=768, CT=1536, Ttf=1024, Ltf=2047, TARGET=896;
  const int F[5] = {768,896,1024,1280,1536};

  // ---- walk inputs in setup_inputs() insertion order ----
  int ix = 0;
  auto nxt = [&]()->const float* { return (const float*)d_in[ix++]; };
  const float* x_in   = nxt();
  const float* stem_w = nxt();
  const float* stem_b = nxt();
  auto getBN = [&](){ BNp r; r.g=nxt(); r.b=nxt(); r.m=nxt(); r.v=nxt(); return r; };
  auto getCB = [&](){ CBp r; r.bn=getBN(); r.w=nxt(); r.bias=nxt(); return r; };
  CBp stem_res = getCB();
  nxt();                                   // stem_pool (identity) — pooling fused elementwise
  CBp tcb[5], tres[5];
  for(int i=0;i<5;++i){ tcb[i]=getCB(); tres[i]=getCB(); nxt(); }   // pool identity
  struct TFp { const float *ln1g,*ln1b,*wq,*wk,*wv,*wr,*rw,*rr,*wo,*bo,*ln2g,*ln2b,*w1,*b1,*w2,*b2; };
  TFp tf[11];
  for(int i=0;i<11;++i){
    tf[i].ln1g=nxt(); tf[i].ln1b=nxt();
    tf[i].wq=nxt(); tf[i].wk=nxt(); tf[i].wv=nxt(); tf[i].wr=nxt();
    tf[i].rw=nxt(); tf[i].rr=nxt(); tf[i].wo=nxt(); tf[i].bo=nxt();
    tf[i].ln2g=nxt(); tf[i].ln2b=nxt();
    tf[i].w1=nxt(); tf[i].b1=nxt(); tf[i].w2=nxt(); tf[i].b2=nxt();
  }
  CBp fin = getCB();
  const float* head_w = nxt();
  const float* head_b = nxt();

  // ---- workspace layout (bump, 256B aligned) ----
  char* wsb = (char*)d_ws;
  const i64 NX = (i64)SEQ*C0;                    // 50,331,648 elts
  i64 off = 0;
  auto take = [&](i64 bytes)->i64 { i64 o=off; off = (off + bytes + 255) & ~(i64)255; return o; };
  i64 OFF_X0   = take(NX*4);
  i64 OFF_X1   = take(NX*4);
  i64 OFF_F32A = take(NX*4);
  const i64 H16E = (i64)32768*3840;              // max im2col elems
  i64 OFF_H16A = take(H16E*2);
  i64 OFF_W16  = take((i64)24<<20);
  i64 OFF_PE   = take((i64)Ltf*192*2);
  i64 OFF_PRB  = take((i64)Ltf*32*4);
  i64 OFF_PMX  = take(256);

  float* X0   = (float*)(wsb+OFF_X0);
  float* X1   = (float*)(wsb+OFF_X1);
  float* F32A = (float*)(wsb+OFF_F32A);
  f16*   H16A = (f16*)(wsb+OFF_H16A);
  f16*   W16  = (f16*)(wsb+OFF_W16);
  f16*   PE16 = (f16*)(wsb+OFF_PE);
  float* PRB  = (float*)(wsb+OFF_PRB);
  float* PMX  = (float*)(wsb+OFF_PMX);

  auto cvt = [&](const float* w, i64 n)->f16* {
    k_f32_to_f16<<<g1(n),256,0,stream>>>(w, W16, n);
    return W16;
  };

  // ---- stem: conv15 4->768, res block (1x1 + BN/GELU), softmax-pool ----
  k_stem_conv<<<g1((i64)SEQ*C0),256,0,stream>>>(x_in, stem_w, stem_b, X0, SEQ, C0);
  k_bn_gelu_f16<<<g1(NX),256,0,stream>>>(X0, stem_res.bn.g,stem_res.bn.b,stem_res.bn.m,stem_res.bn.v, H16A, NX, C0);
  { f16* w = cvt(stem_res.w, (i64)C0*C0);
    gemm(stream, H16A, C0,0, w, C0,0,0, stem_res.bias, X0,0, X1,nullptr, C0,0, SEQ, C0, C0, 1.f, 0, 1); }
  k_pool2<<<g1((i64)(SEQ/2)*C0),256,0,stream>>>(X1, X0, SEQ/2, C0);

  // ---- tower ----
  int T = SEQ/2, cin = C0;
  for(int s=0;s<5;++s){
    int f = F[s];
    int Kk = 5*cin;
    k_im2col_bn_gelu<<<g1((i64)T*Kk),256,0,stream>>>(X0, tcb[s].bn.g,tcb[s].bn.b,tcb[s].bn.m,tcb[s].bn.v, H16A, T, cin);
    { f16* w = cvt(tcb[s].w, (i64)Kk*f);
      gemm(stream, H16A, Kk,0, w, f,0,0, tcb[s].bias, nullptr,0, F32A,nullptr, f,0, T, f, Kk, 1.f, 0, 1); }
    k_bn_gelu_f16<<<g1((i64)T*f),256,0,stream>>>(F32A, tres[s].bn.g,tres[s].bn.b,tres[s].bn.m,tres[s].bn.v, H16A, (i64)T*f, f);
    { f16* w = cvt(tres[s].w, (i64)f*f);
      gemm(stream, H16A, f,0, w, f,0,0, tres[s].bias, F32A,0, X1,nullptr, f,0, T, f, f, 1.f, 0, 1); }
    k_pool2<<<g1((i64)(T/2)*f),256,0,stream>>>(X1, X0, T/2, f);
    T >>= 1; cin = f;
  }
  // now X0 = x [1024,1536]

  // ---- relative position features (once) ----
  k_zero1<<<1,256,0,stream>>>(PMX);
  k_pe_prob<<<g1((i64)Ltf*32),256,0,stream>>>(PRB, PMX, Ttf);
  k_pe_emb<<<g1((i64)Ltf*192),256,0,stream>>>(PRB, PMX, PE16, Ttf);

  // ---- transformer sub-buffers ----
  float* qs      = F32A;                                   // [1024,512]
  float* content = (float*)((char*)F32A + ((i64)4<<20));   // [8,1024,1024]
  float* rel     = (float*)((char*)F32A + ((i64)40<<20));  // [8,1024,2047]
  f16* ln16  = H16A;                                       // [1024,1536]
  f16* qrw   = (f16*)((char*)H16A + ((i64)4<<20));
  f16* qrr   = (f16*)((char*)H16A + ((i64)6<<20));
  f16* kk16  = (f16*)((char*)H16A + ((i64)8<<20));
  f16* vv16  = (f16*)((char*)H16A + ((i64)10<<20));
  f16* rk16  = (f16*)((char*)H16A + ((i64)14<<20));
  f16* pr16  = (f16*)((char*)H16A + ((i64)18<<20));        // probs [8,1024,1024]
  f16* ao16  = (f16*)((char*)H16A + ((i64)36<<20));        // attn out [1024,1536]
  f16* h116  = (f16*)((char*)H16A + ((i64)40<<20));        // ffn hidden [1024,3072]

  const i64 TT = (i64)Ttf*Ttf;
  for(int l=0;l<11;++l){
    const TFp& p = tf[l];
    // ln1
    k_layernorm_f16<<<Ttf,256,0,stream>>>(X0, p.ln1g, p.ln1b, ln16, CT);
    // q (scaled), +r_w / +r_r
    { f16* w = cvt(p.wq, (i64)CT*512);
      gemm(stream, ln16, CT,0, w, 512,0,0, nullptr, nullptr,0, qs,nullptr, 512,0, Ttf, 512, CT, 0.125f, 0, 1); }
    k_add_head_bias<<<g1((i64)Ttf*512),256,0,stream>>>(qs, p.rw, qrw, (i64)Ttf*512);
    k_add_head_bias<<<g1((i64)Ttf*512),256,0,stream>>>(qs, p.rr, qrr, (i64)Ttf*512);
    // k, v
    { f16* w = cvt(p.wk, (i64)CT*512);
      gemm(stream, ln16, CT,0, w, 512,0,0, nullptr, nullptr,0, nullptr,kk16, 512,0, Ttf, 512, CT, 1.f, 0, 1); }
    { f16* w = cvt(p.wv, (i64)CT*1536);
      gemm(stream, ln16, CT,0, w, 1536,0,0, nullptr, nullptr,0, nullptr,vv16, 1536,0, Ttf, 1536, CT, 1.f, 0, 1); }
    // r_k = pe @ wr
    { f16* w = cvt(p.wr, (i64)192*512);
      gemm(stream, PE16, 192,0, w, 512,0,0, nullptr, nullptr,0, nullptr,rk16, 512,0, Ltf, 512, 192, 1.f, 0, 1); }
    // content[h] = (q+r_w)_h @ k_h^T     (batched over heads)
    gemm(stream, qrw, 512,64, kk16, 512,64, 1, nullptr, nullptr,0,
         content,nullptr, Ttf, TT, Ttf, Ttf, 64, 1.f, 0, 8);
    // rel[h] = (q+r_r)_h @ r_k_h^T
    gemm(stream, qrr, 512,64, rk16, 512,64, 1, nullptr, nullptr,0,
         rel,nullptr, Ltf, (i64)Ttf*Ltf, Ttf, Ltf, 64, 1.f, 0, 8);
    // softmax(content + shift(rel))
    { dim3 grid(Ttf, 8); k_attn_softmax<<<grid,256,0,stream>>>(content, rel, pr16, Ttf); }
    // out_h = probs_h @ v_h
    gemm(stream, pr16, Ttf, TT, vv16, 1536,192, 0, nullptr, nullptr,0,
         nullptr, ao16, 1536, 192, Ttf, 192, Ttf, 1.f, 0, 8);
    // x = x + out @ wo + bo
    { f16* w = cvt(p.wo, (i64)1536*CT);
      gemm(stream, ao16, 1536,0, w, CT,0,0, p.bo, X0,0, X1,nullptr, CT,0, Ttf, CT, 1536, 1.f, 0, 1); }
    // ffn
    k_layernorm_f16<<<Ttf,256,0,stream>>>(X1, p.ln2g, p.ln2b, ln16, CT);
    { f16* w = cvt(p.w1, (i64)CT*3072);
      gemm(stream, ln16, CT,0, w, 3072,0,0, p.b1, nullptr,0, nullptr,h116, 3072,0, Ttf, 3072, CT, 1.f, 1, 1); }
    { f16* w = cvt(p.w2, (i64)3072*CT);
      gemm(stream, h116, 3072,0, w, CT,0,0, p.b2, X1,0, X0,nullptr, CT,0, Ttf, CT, 3072, 1.f, 0, 1); }
  }

  // ---- final: trim, conv_block (1x1) with outer GELU, head + softplus ----
  const float* xtrim = X0 + (i64)64*CT;   // rows 64..959
  k_bn_gelu_f16<<<g1((i64)TARGET*CT),256,0,stream>>>(xtrim, fin.bn.g,fin.bn.b,fin.bn.m,fin.bn.v, H16A, (i64)TARGET*CT, CT);
  f16* fo16 = (f16*)((char*)H16A + ((i64)8<<20));   // [896,3072]
  { f16* w = cvt(fin.w, (i64)CT*3072);
    gemm(stream, H16A, CT,0, w, 3072,0,0, fin.bias, nullptr,0, nullptr,fo16, 3072,0, TARGET, 3072, CT, 1.f, 2, 1); }
  { f16* w = cvt(head_w, (i64)3072*2);
    gemm(stream, fo16, 3072,0, w, 2,0,0, head_b, nullptr,0, (float*)d_out,nullptr, 2,0, TARGET, 2, 3072, 1.f, 3, 1); }
}
